// TarsMamba2LM_55336358642849
// MI455X (gfx1250) — matrix operand order repaired and checked by reading, not compiled, not run on hardware
//
#include <hip/hip_runtime.h>

// Problem constants (reference: B=8, L=4096, D=2048)
#define D_MODEL 2048
#define SEQ_L   4096
#define BATCH   8
#define NCHUNK  32                       // L-chunks for the mean reduction
#define ROWS_PER_CHUNK (SEQ_L / NCHUNK)  // 128
#define H_K     (2 * D_MODEL)            // 4096 (concat width)

typedef __attribute__((ext_vector_type(16))) _Float16 v16h;
typedef __attribute__((ext_vector_type(8)))  float    v8f;

enum { ACT_NONE = 0, ACT_SIGMOID = 1, ACT_SILU = 2, ACT_TANH = 3 };

__device__ __forceinline__ float act_apply(float x, int act) {
  switch (act) {
    case ACT_SIGMOID: return 1.0f / (1.0f + __expf(-x));
    case ACT_SILU:    return x / (1.0f + __expf(-x));
    case ACT_TANH:    return tanhf(x);
    default:          return x;
  }
}

// ---------------------------------------------------------------------------
// Stage 1: partial column sums over L (deterministic two-stage reduction).
// grid = (D_MODEL/256, NCHUNK, 2*BATCH), block = 256
// ---------------------------------------------------------------------------
__global__ void col_partial_sums(const float* __restrict__ xl,
                                 const float* __restrict__ xr,
                                 float* __restrict__ partials) {
  const int col = blockIdx.x * blockDim.x + threadIdx.x;   // 0..2047
  const int t   = blockIdx.z >> 3;                         // 0 = left, 1 = right
  const int b   = blockIdx.z & 7;
  const float* __restrict__ src = (t == 0 ? xl : xr)
      + ((size_t)b * SEQ_L + (size_t)blockIdx.y * ROWS_PER_CHUNK) * D_MODEL + col;
  float s = 0.0f;
  #pragma unroll 4
  for (int i = 0; i < ROWS_PER_CHUNK; ++i) {
    // speculative prefetch ahead of the 8KB-stride stream (global_prefetch_b8)
    __builtin_prefetch(src + (size_t)(i + 24) * D_MODEL, 0, 0);
    s += src[(size_t)i * D_MODEL];
  }
  partials[((size_t)blockIdx.z * NCHUNK + blockIdx.y) * D_MODEL + col] = s;
}

// ---------------------------------------------------------------------------
// Stage 2: reduce chunk partials -> h_cat, convert to f16, pad M 8->16 with 0.
// h16 layout: 16 rows x 4096 cols, row b = [mean_left(b) | mean_right(b)].
// grid = (16*H_K/256), block = 256
// ---------------------------------------------------------------------------
__global__ void finalize_h16(const float* __restrict__ partials,
                             _Float16* __restrict__ h16) {
  const int idx = blockIdx.x * blockDim.x + threadIdx.x;   // 0 .. 16*4096-1
  const int row = idx >> 12;            // /4096
  const int col = idx & (H_K - 1);
  float v = 0.0f;
  if (row < BATCH) {
    const int t = col >> 11;            // which source tensor
    const int c = col & (D_MODEL - 1);
    const int z = t * BATCH + row;
    const float* __restrict__ p = partials + (size_t)z * NCHUNK * D_MODEL + c;
    #pragma unroll
    for (int i = 0; i < NCHUNK; ++i) v += p[(size_t)i * D_MODEL];
    v *= (1.0f / (float)SEQ_L);
  }
  h16[idx] = (_Float16)v;               // rows 8..15 stay exactly zero
}

// ---------------------------------------------------------------------------
// WMMA GEMM: out[16, N] = act(A16[16,K] @ W[N,K]^T + bias), one wave per
// 16x16 output tile, K-loop of v_wmma_f32_16x16x32_f16 with f32 accumulate.
// Weights are streamed fp32 and converted to f16 in registers.
// grid = (N/128), block = 256 (8 waves x 16 cols)
// ---------------------------------------------------------------------------
template <int ACT, bool WF32, bool WF16>
__global__ void gemm16_wmma(const _Float16* __restrict__ A16,  // 16 x K row-major
                            const float* __restrict__ W,       // N x K row-major
                            const float* __restrict__ bias,    // >= N entries
                            float* __restrict__ outF,          // 8 x N (rows 0..7)
                            _Float16* __restrict__ outH,       // 16 x N (padded)
                            int K, int N) {
  const int lane   = threadIdx.x & 31;
  const int wave   = threadIdx.x >> 5;
  const int n0     = (blockIdx.x * (blockDim.x >> 5) + wave) * 16;
  if (n0 >= N) return;                    // uniform per wave: EXEC stays all-ones
  const int r      = lane & 15;           // A row (M) / B column (N) in tile
  const int hiHalf = lane >> 4;           // 0 or 1
  const int koffA  = hiHalf * 8;
  const int koffB  = hiHalf * 16;

  const _Float16* __restrict__ arow = A16 + (size_t)r * K;
  const float*    __restrict__ wrow = W   + (size_t)(n0 + r) * K;

  v8f acc = {};
  for (int k0 = 0; k0 < K; k0 += 32) {
    // A fragment: lanes 0-15 -> K {0..15}; lanes 16-31 -> K {16..31}
    v16h a;
    #pragma unroll
    for (int j = 0; j < 8; ++j) {
      a[j]     = arow[k0 + koffB + j];
      a[8 + j] = arow[k0 + koffB + 8 + j];
    }
    // B fragment: lane holds 16 contiguous K values of column n0+r (W row-major)
    v16h bm;
    const float4* __restrict__ w4 = (const float4*)(wrow + k0 + koffB);
    #pragma unroll
    for (int q = 0; q < 4; ++q) {
      float4 wq = w4[q];
      bm[4 * q + 0] = (_Float16)wq.x;
      bm[4 * q + 1] = (_Float16)wq.y;
      bm[4 * q + 2] = (_Float16)wq.z;
      bm[4 * q + 3] = (_Float16)wq.w;
    }
    acc = __builtin_amdgcn_wmma_f32_16x16x32_f16(
        /*neg_a=*/false, a, /*neg_b=*/false, bm,
        /*c_mod=*/(short)0, acc, /*reuse_a=*/false, /*reuse_b=*/false);
  }

  // C/D layout: VGPR v -> M = hiHalf*8 + v, N = n0 + (lane&15)
  const int   ncol  = n0 + r;
  const int   mbase = hiHalf * 8;
  const float bv    = bias[ncol];
  #pragma unroll
  for (int v = 0; v < 8; ++v) {
    const int m = mbase + v;
    const float val = act_apply(acc[v] + bv, ACT);
    if (WF32) { if (m < BATCH) outF[(size_t)m * N + ncol] = val; }
    if (WF16) { outH[(size_t)m * N + ncol] = (m < BATCH) ? (_Float16)val
                                                         : (_Float16)0.0f; }
  }
}

// ---------------------------------------------------------------------------
// Stage 3: fused gate + fusion/reflex broadcast + LayerNorm, one row per block.
// grid = (BATCH*SEQ_L), block = 256, 8 cols per thread held in registers.
// ---------------------------------------------------------------------------
__global__ void gate_fuse_ln(const float* __restrict__ xl,
                             const float* __restrict__ xr,
                             const float* __restrict__ alpha,
                             const float* __restrict__ fusion,
                             const float* __restrict__ reflex,
                             const float* __restrict__ gamma,
                             const float* __restrict__ beta,
                             const float* __restrict__ fs_p,
                             const float* __restrict__ rs_p,
                             float* __restrict__ out) {
  const int row = blockIdx.x;            // b * SEQ_L + l
  const int b   = row >> 12;             // / SEQ_L
  const float fs = fs_p[0], rs = rs_p[0];
  const float* __restrict__ xlr = xl + (size_t)row * D_MODEL;
  const float* __restrict__ xrr = xr + (size_t)row * D_MODEL;
  const float* __restrict__ ab  = alpha  + b * D_MODEL;
  const float* __restrict__ fb  = fusion + b * D_MODEL;
  const float* __restrict__ rb  = reflex + b * D_MODEL;

  float v[8];
  float sum = 0.0f, ssq = 0.0f;
  #pragma unroll
  for (int i = 0; i < 8; ++i) {
    const int j = threadIdx.x + i * 256;
    const float a = ab[j];
    const float x = (1.0f - a) * xlr[j] + a * xrr[j] + fs * fb[j] + rs * rb[j];
    v[i] = x;
    sum += x;
    ssq += x * x;
  }
  // wave32 shuffle reduce, then cross-wave via LDS
  #pragma unroll
  for (int off = 16; off > 0; off >>= 1) {
    sum += __shfl_down(sum, off, 32);
    ssq += __shfl_down(ssq, off, 32);
  }
  __shared__ float sA[8], sB[8];
  const int lane = threadIdx.x & 31, wave = threadIdx.x >> 5;
  if (lane == 0) { sA[wave] = sum; sB[wave] = ssq; }
  __syncthreads();
  if (threadIdx.x == 0) {
    float a = 0.0f, c = 0.0f;
    #pragma unroll
    for (int i = 0; i < 8; ++i) { a += sA[i]; c += sB[i]; }
    sA[0] = a; sB[0] = c;
  }
  __syncthreads();
  const float mu  = sA[0] * (1.0f / (float)D_MODEL);
  const float var = sB[0] * (1.0f / (float)D_MODEL) - mu * mu;
  const float inv = rsqrtf(var + 1e-5f);
  float* __restrict__ o = out + (size_t)row * D_MODEL;
  #pragma unroll
  for (int i = 0; i < 8; ++i) {
    const int j = threadIdx.x + i * 256;
    o[j] = (v[i] - mu) * inv * gamma[j] + beta[j];
  }
}

// ---------------------------------------------------------------------------
// alpha.mean() -> single scalar at the tail of d_out
// ---------------------------------------------------------------------------
__global__ void alpha_mean_kernel(const float* __restrict__ alpha,
                                  float* __restrict__ out_scalar) {
  float s = 0.0f;
  for (int i = threadIdx.x; i < BATCH * D_MODEL; i += 256) s += alpha[i];
  #pragma unroll
  for (int off = 16; off > 0; off >>= 1) s += __shfl_down(s, off, 32);
  __shared__ float sA[8];
  const int lane = threadIdx.x & 31, wave = threadIdx.x >> 5;
  if (lane == 0) sA[wave] = s;
  __syncthreads();
  if (threadIdx.x == 0) {
    float t = 0.0f;
    #pragma unroll
    for (int i = 0; i < 8; ++i) t += sA[i];
    out_scalar[0] = t * (1.0f / (float)(BATCH * D_MODEL));
  }
}

// ---------------------------------------------------------------------------
extern "C" void kernel_launch(void* const* d_in, const int* in_sizes, int n_in,
                              void* d_out, int out_size, void* d_ws, size_t ws_size,
                              hipStream_t stream) {
  (void)in_sizes; (void)n_in; (void)out_size; (void)ws_size;
  const float* xl     = (const float*)d_in[0];
  const float* xr     = (const float*)d_in[1];
  const float* W_gate = (const float*)d_in[2];
  const float* b_gate = (const float*)d_in[3];
  const float* W_f1   = (const float*)d_in[4];
  const float* b_f1   = (const float*)d_in[5];
  const float* W_f2   = (const float*)d_in[6];
  const float* b_f2   = (const float*)d_in[7];   // zeros (reference shape quirk)
  const float* W_f3   = (const float*)d_in[8];
  const float* b_f3   = (const float*)d_in[9];
  const float* W_r    = (const float*)d_in[10];
  const float* b_r    = (const float*)d_in[11];
  const float* gamma  = (const float*)d_in[12];
  const float* beta   = (const float*)d_in[13];
  const float* fs     = (const float*)d_in[14];
  const float* rs     = (const float*)d_in[15];
  float* out = (float*)d_out;

  // Workspace carve-up (~4.7 MB total)
  char* ws = (char*)d_ws;
  float*    partials = (float*)ws;    ws += (size_t)2 * BATCH * NCHUNK * D_MODEL * 4;
  _Float16* h16      = (_Float16*)ws; ws += (size_t)16 * H_K * 2;
  _Float16* f1_16    = (_Float16*)ws; ws += (size_t)16 * H_K * 2;
  _Float16* f2_16    = (_Float16*)ws; ws += (size_t)16 * D_MODEL * 2;
  float*    alphaF   = (float*)ws;    ws += (size_t)BATCH * D_MODEL * 4;
  float*    fusionF  = (float*)ws;    ws += (size_t)BATCH * D_MODEL * 4;
  float*    reflexF  = (float*)ws;

  // 1) column means (two-stage, deterministic)
  col_partial_sums<<<dim3(D_MODEL / 256, NCHUNK, 2 * BATCH), 256, 0, stream>>>(
      xl, xr, partials);
  finalize_h16<<<dim3(16 * H_K / 256), 256, 0, stream>>>(partials, h16);

  // 2) WMMA linears (M=8 padded to one 16-row tile)
  gemm16_wmma<ACT_SIGMOID, true,  false><<<dim3(D_MODEL / 128), 256, 0, stream>>>(
      h16, W_gate, b_gate, alphaF, nullptr, H_K, D_MODEL);
  gemm16_wmma<ACT_SILU,    false, true ><<<dim3(H_K / 128),     256, 0, stream>>>(
      h16, W_f1, b_f1, nullptr, f1_16, H_K, H_K);
  gemm16_wmma<ACT_TANH,    true,  false><<<dim3(D_MODEL / 128), 256, 0, stream>>>(
      h16, W_r, b_r, reflexF, nullptr, H_K, D_MODEL);
  gemm16_wmma<ACT_SILU,    false, true ><<<dim3(D_MODEL / 128), 256, 0, stream>>>(
      f1_16, W_f2, b_f2, nullptr, f2_16, H_K, D_MODEL);
  gemm16_wmma<ACT_NONE,    true,  false><<<dim3(D_MODEL / 128), 256, 0, stream>>>(
      f2_16, W_f3, b_f3, fusionF, nullptr, D_MODEL, D_MODEL);

  // 3) fused gate + broadcast-add + LayerNorm (the one big HBM sweep)
  gate_fuse_ln<<<dim3(BATCH * SEQ_L), 256, 0, stream>>>(
      xl, xr, alphaF, fusionF, reflexF, gamma, beta, fs, rs, out);

  // 4) alpha.mean() scalar appended after x_norm
  alpha_mean_kernel<<<dim3(1), 256, 0, stream>>>(
      alphaF, out + (size_t)BATCH * SEQ_L * D_MODEL);
}
